// Attention_41102837023186
// MI455X (gfx1250) — matrix operand order repaired and checked by reading, not compiled, not run on hardware
//
#include <hip/hip_runtime.h>
#include <hip/hip_bf16.h>

// ---------------------------------------------------------------------------
// Problem constants (from reference)
// ---------------------------------------------------------------------------
constexpr int S   = 3072;
constexpr int H   = 3072;
constexpr int HQ  = 24;
constexpr int HKV = 8;
constexpr int D   = 128;
constexpr int REP = HQ / HKV;     // 3
constexpr int BLK = S / 4;        // 768 : query block i attends keys [0,(i+1)*768)

typedef __attribute__((ext_vector_type(16))) __bf16 v16bf;
typedef __attribute__((ext_vector_type(8)))  __bf16 v8bf;
typedef __attribute__((ext_vector_type(8)))  float  v8f;
typedef __attribute__((ext_vector_type(4)))  unsigned v4u;
typedef __attribute__((ext_vector_type(8)))  unsigned v8u;

union ABfrag { v16bf v; v8bf h[2]; };

// ---------------------------------------------------------------------------
// TDM: issue a 2D tensor_load_to_lds of tileRows x 32 bf16 elements, rows
// strided by strideElems in global memory, compacted into LDS with 16B pad
// after every 64B row (-> 80B LDS row stride, conflict-friendly).
// Descriptor per CDNA5 ISA. Wave-level op, EXEC ignored.
// ---------------------------------------------------------------------------
__device__ __forceinline__ void tdm_load_tile(unsigned ldsOff, const __bf16* g,
                                              unsigned tileRows, unsigned strideElems) {
  unsigned long long ga = (unsigned long long)(const void*)g;
  v4u g0;
  g0.x = 1u;                                                   // count=1, user mode
  g0.y = ldsOff;                                               // lds_addr (bytes)
  g0.z = (unsigned)ga;                                         // global_addr[31:0]
  g0.w = (unsigned)((ga >> 32) & 0x1FFFFFFull) | (2u << 30);   // addr[56:32] | type=2
  v8u g1;
  g1[0] = (1u << 16)        // data_size = 1 -> 2 bytes (bf16)
        | (1u << 20)        // pad_enable
        | (3u << 22)        // pad_interval = 16 DWORDs (after every 64B stored)
        | (3u << 25);       // pad_amount   = 4 DWORDs (16B)
  g1[1] = 32u << 16;                 // tensor_dim0 = 32 elems (tile never OOB)
  g1[2] = tileRows << 16;            // tensor_dim1[15:0]
  g1[3] = 32u << 16;                 // tile_dim0 = 32 elems
  g1[4] = tileRows;                  // tile_dim1 (tile_dim2 = 0)
  g1[5] = strideElems;               // tensor_dim0_stride[31:0]
  g1[6] = 0u;                        // stride high bits / dim1_stride unused (2D)
  g1[7] = 0u;
  asm volatile("tensor_load_to_lds %0, %1" :: "s"(g0), "s"(g1) : "memory");
}

// ---------------------------------------------------------------------------
// fp32 -> bf16 elementwise convert
// ---------------------------------------------------------------------------
__global__ __launch_bounds__(256) void convert_f32_bf16(__bf16* __restrict__ dst,
                                                        const float* __restrict__ src,
                                                        int n) {
  int i = blockIdx.x * 256 + threadIdx.x;
  if (i < n) dst[i] = (__bf16)src[i];
}

// ---------------------------------------------------------------------------
// Row RMSNorm: out_bf16[s][i] = x*rsqrt(mean(x^2)+eps)*(w+1)
// ---------------------------------------------------------------------------
__global__ __launch_bounds__(256) void rmsnorm_rows(__bf16* __restrict__ out,
                                                    const float* __restrict__ x,
                                                    const float* __restrict__ w) {
  const int s = blockIdx.x;
  const float* row = x + (long)s * H;
  float ss = 0.f;
  for (int i = threadIdx.x; i < H; i += 256) { float v = row[i]; ss += v * v; }
  __shared__ float red[256];
  red[threadIdx.x] = ss;
  __syncthreads();
  for (int off = 128; off; off >>= 1) {
    if (threadIdx.x < off) red[threadIdx.x] += red[threadIdx.x + off];
    __syncthreads();
  }
  const float inv = rsqrtf(red[0] / (float)H + 1e-6f);
  for (int i = threadIdx.x; i < H; i += 256)
    out[(long)s * H + i] = (__bf16)(row[i] * inv * (w[i] + 1.0f));
}

// ---------------------------------------------------------------------------
// TDM-staged WMMA GEMM: C[M,N] = A[M,K] @ B[N,K]^T
// Block = 8 waves, macro tile 64(M) x 128(N); wave = 32x32 (2x2 16x16 WMMA).
// Per 32-wide K step, wave 0 DMAs A(64x32) and B(128x32) bf16 slices into LDS
// via tensor_load_to_lds (double-buffered, s_wait_tensorcnt + barrier); all
// waves read WMMA fragments from LDS with ds_load_b128 (80B padded rows).
// ---------------------------------------------------------------------------
template <bool STORE_BF16>
__global__ __launch_bounds__(256) void gemm_bf16_tdm(void* __restrict__ Cv,
                                                     const __bf16* __restrict__ A,
                                                     const __bf16* __restrict__ B,
                                                     int M, int N, int K) {
  constexpr int LDR = 40;                       // LDS row stride in bf16 (80B, TDM-padded)
  __shared__ __bf16 ldsA[2][64 * LDR];
  __shared__ __bf16 ldsB[2][128 * LDR];

  const int lane = threadIdx.x & 31;
  const int wave = threadIdx.x >> 5;
  const int ln = lane & 15, hf = lane >> 4;
  const int tilesN = N / 128;
  const int bm = blockIdx.x / tilesN, bn = blockIdx.x % tilesN;
  const int wm = wave >> 2, wn = wave & 3;      // 2 x 4 wave grid

  const __bf16* aBase = A + (long)bm * 64 * K;
  const __bf16* bBase = B + (long)bn * 128 * K;
  const unsigned offA0 = (unsigned)(unsigned long long)(const void*)&ldsA[0][0];
  const unsigned offA1 = (unsigned)(unsigned long long)(const void*)&ldsA[1][0];
  const unsigned offB0 = (unsigned)(unsigned long long)(const void*)&ldsB[0][0];
  const unsigned offB1 = (unsigned)(unsigned long long)(const void*)&ldsB[1][0];

  const int kSteps = K / 32;
  if (wave == 0) {                              // prologue: fill buffer 0
    tdm_load_tile(offA0, aBase, 64u, (unsigned)K);
    tdm_load_tile(offB0, bBase, 128u, (unsigned)K);
  }

  v8f acc[2][2] = {};
  for (int i = 0; i < kSteps; ++i) {
    const int buf = i & 1;
    if (wave == 0) {
      if (i + 1 < kSteps) {                     // issue next slice into other buffer
        tdm_load_tile(buf ? offA0 : offA1, aBase + (long)(i + 1) * 32, 64u, (unsigned)K);
        tdm_load_tile(buf ? offB0 : offB1, bBase + (long)(i + 1) * 32, 128u, (unsigned)K);
        __builtin_amdgcn_s_wait_tensorcnt(2);   // in-order: current buffer complete
      } else {
        __builtin_amdgcn_s_wait_tensorcnt(0);
      }
    }
    __syncthreads();                            // data visible to all 8 waves

    ABfrag af[2], bf[2];
#pragma unroll
    for (int mi = 0; mi < 2; ++mi) {
      const __bf16* ar = &ldsA[buf][(wm * 32 + mi * 16 + ln) * LDR];
      af[mi].h[0] = *(const v8bf*)(ar + hf * 8);
      af[mi].h[1] = *(const v8bf*)(ar + 16 + hf * 8);
    }
#pragma unroll
    for (int nj = 0; nj < 2; ++nj) {
      const __bf16* br = &ldsB[buf][(wn * 32 + nj * 16 + ln) * LDR];
      bf[nj].h[0] = *(const v8bf*)(br + hf * 8);
      bf[nj].h[1] = *(const v8bf*)(br + 16 + hf * 8);
    }
#pragma unroll
    for (int mi = 0; mi < 2; ++mi)
#pragma unroll
      for (int nj = 0; nj < 2; ++nj)
        acc[mi][nj] = __builtin_amdgcn_wmma_f32_16x16x32_bf16(
            false, af[mi].v, false, bf[nj].v, (short)0, acc[mi][nj], false, false);

    __syncthreads();                            // done reading before next overwrite
  }

#pragma unroll
  for (int mi = 0; mi < 2; ++mi)
#pragma unroll
    for (int nj = 0; nj < 2; ++nj) {
      const int col = bn * 128 + wn * 32 + nj * 16 + ln;
#pragma unroll
      for (int v = 0; v < 8; ++v) {
        const int row = bm * 64 + wm * 32 + mi * 16 + v + 8 * hf;
        if (STORE_BF16) ((__bf16*)Cv)[(long)row * N + col] = (__bf16)acc[mi][nj][v];
        else            ((float*)Cv)[(long)row * N + col]  = acc[mi][nj][v];
      }
    }
}

// ---------------------------------------------------------------------------
// In-place per-head RMSNorm + RoPE on bf16 q/k.  One wave per (s, head).
// ---------------------------------------------------------------------------
__global__ __launch_bounds__(256) void rope_norm_inplace(__bf16* __restrict__ q,
                                                         const float* __restrict__ cos_,
                                                         const float* __restrict__ sin_,
                                                         const float* __restrict__ w,
                                                         int nh) {
  const int wid  = blockIdx.x * 8 + (threadIdx.x >> 5);
  const int lane = threadIdx.x & 31;
  const int s = wid / nh, h = wid % nh;
  __bf16* row = q + ((long)s * nh + h) * D;

  float x0 = (float)row[lane],      x1 = (float)row[lane + 32];
  float x2 = (float)row[lane + 64], x3 = (float)row[lane + 96];
  float ss = x0 * x0 + x1 * x1 + x2 * x2 + x3 * x3;
#pragma unroll
  for (int off = 16; off; off >>= 1) ss += __shfl_xor(ss, off, 32);
  const float inv = rsqrtf(ss / (float)D + 1e-6f);
  x0 *= inv * (w[lane] + 1.f);      x1 *= inv * (w[lane + 32] + 1.f);
  x2 *= inv * (w[lane + 64] + 1.f); x3 *= inv * (w[lane + 96] + 1.f);

  const float c0 = cos_[s * 64 + lane],      s0 = sin_[s * 64 + lane];
  const float c1 = cos_[s * 64 + lane + 32], s1 = sin_[s * 64 + lane + 32];
  row[lane]      = (__bf16)(x0 * c0 - x2 * s0);
  row[lane + 64] = (__bf16)(x2 * c0 + x0 * s0);
  row[lane + 32] = (__bf16)(x1 * c1 - x3 * s1);
  row[lane + 96] = (__bf16)(x3 * c1 + x1 * s1);
}

// ---------------------------------------------------------------------------
// V transpose: vt[(h*D+d)*S + s] = v[s][h*D+d]
// ---------------------------------------------------------------------------
__global__ __launch_bounds__(256) void transpose_v(__bf16* __restrict__ vt,
                                                   const __bf16* __restrict__ vsrc) {
  const long i  = (long)blockIdx.x * 256 + threadIdx.x;
  const int  s  = (int)(i / (HKV * D));
  const int  hd = (int)(i % (HKV * D));
  vt[(long)hd * S + s] = vsrc[i];
}

// ---------------------------------------------------------------------------
// Flash attention, one wave per (head, 16-query tile).  (unchanged layout)
// ---------------------------------------------------------------------------
__global__ __launch_bounds__(256) void attn_wmma(__bf16* __restrict__ out,
                                                 const __bf16* __restrict__ Q,
                                                 const __bf16* __restrict__ Km,
                                                 const __bf16* __restrict__ Vt) {
  const int lane = threadIdx.x & 31;
  const int wave = threadIdx.x >> 5;
  const int ln = lane & 15, hf = lane >> 4;
  const int wid = blockIdx.x * 8 + wave;
  const int h  = wid / (S / 16);
  const int qt = wid % (S / 16);
  const int kvh = h / REP;

  __shared__ __bf16 plds[8][16][40];

  ABfrag qf[4];
  const __bf16* qRow = Q + ((long)(qt * 16 + ln) * HQ + h) * D;
#pragma unroll
  for (int c = 0; c < 4; ++c) {
    qf[c].h[0] = *(const v8bf*)(qRow + c * 32 + hf * 8);
    qf[c].h[1] = *(const v8bf*)(qRow + c * 32 + 16 + hf * 8);
  }

  v8f o[8] = {};
  float mrow[8], lrow[8];
#pragma unroll
  for (int v = 0; v < 8; ++v) { mrow[v] = -1e30f; lrow[v] = 0.f; }

  const int klimit = ((qt * 16) / BLK + 1) * BLK;
  const float scale = 0.08838834764831845f;

  for (int kb = 0; kb < klimit; kb += 32) {
    // prefetch next key block's K rows and Vt rows (-> global_prefetch_b8)
    if (kb + 32 < klimit) {
      __builtin_prefetch(Km + ((long)(kb + 32 + lane) * HKV + kvh) * D, 0, 1);
      __builtin_prefetch(Vt + ((long)kvh * D + lane * 4) * S + kb + 32, 0, 1);
    }
    // ---- S = Q @ K^T ----
    v8f sc[2] = {};
#pragma unroll
    for (int t = 0; t < 2; ++t) {
#pragma unroll
      for (int c = 0; c < 4; ++c) {
        const __bf16* kRow = Km + ((long)(kb + t * 16 + ln) * HKV + kvh) * D + c * 32;
        ABfrag bf;
        bf.h[0] = *(const v8bf*)(kRow + hf * 8);
        bf.h[1] = *(const v8bf*)(kRow + 16 + hf * 8);
        sc[t] = __builtin_amdgcn_wmma_f32_16x16x32_bf16(false, qf[c].v, false, bf.v,
                                                        (short)0, sc[t], false, false);
      }
    }
    // ---- online softmax ----
    float alpha[8];
#pragma unroll
    for (int v = 0; v < 8; ++v) {
      float mx = fmaxf(sc[0][v], sc[1][v]) * scale;
#pragma unroll
      for (int off = 8; off; off >>= 1) mx = fmaxf(mx, __shfl_xor(mx, off, 16));
      const float mn = fmaxf(mrow[v], mx);
      alpha[v] = __expf(mrow[v] - mn);
      const float p0 = __expf(sc[0][v] * scale - mn);
      const float p1 = __expf(sc[1][v] * scale - mn);
      sc[0][v] = p0; sc[1][v] = p1;
      float rs = p0 + p1;
#pragma unroll
      for (int off = 8; off; off >>= 1) rs += __shfl_xor(rs, off, 16);
      lrow[v] = lrow[v] * alpha[v] + rs;
      mrow[v] = mn;
    }
    // ---- C-layout P -> A-layout P via per-wave LDS tile ----
#pragma unroll
    for (int v = 0; v < 8; ++v) {
      plds[wave][v + 8 * hf][ln]      = (__bf16)sc[0][v];
      plds[wave][v + 8 * hf][16 + ln] = (__bf16)sc[1][v];
    }
    __builtin_amdgcn_wave_barrier();
    asm volatile("s_wait_dscnt 0" ::: "memory");   // intra-wave LDS RAW fence
#pragma unroll
    for (int dt = 0; dt < 8; ++dt)
#pragma unroll
      for (int v = 0; v < 8; ++v) o[dt][v] *= alpha[v];

    ABfrag pf;
    pf.h[0] = *(const v8bf*)(&plds[wave][ln][hf * 8]);
    pf.h[1] = *(const v8bf*)(&plds[wave][ln][16 + hf * 8]);
    __builtin_amdgcn_wave_barrier();

    // ---- O += P @ V ----
#pragma unroll
    for (int dt = 0; dt < 8; ++dt) {
      const __bf16* vRow = Vt + ((long)kvh * D + dt * 16 + ln) * S + kb;
      ABfrag bf;
      bf.h[0] = *(const v8bf*)(vRow + hf * 8);
      bf.h[1] = *(const v8bf*)(vRow + 16 + hf * 8);
      o[dt] = __builtin_amdgcn_wmma_f32_16x16x32_bf16(false, pf.v, false, bf.v,
                                                      (short)0, o[dt], false, false);
    }
  }
#pragma unroll
  for (int dt = 0; dt < 8; ++dt)
#pragma unroll
    for (int v = 0; v < 8; ++v) {
      const int row = qt * 16 + v + 8 * hf;
      out[((long)row * HQ + h) * D + dt * 16 + ln] = (__bf16)(o[dt][v] / lrow[v]);
    }
}

// ---------------------------------------------------------------------------
// Host-side orchestration
// ---------------------------------------------------------------------------
extern "C" void kernel_launch(void* const* d_in, const int* in_sizes, int n_in,
                              void* d_out, int out_size, void* d_ws, size_t ws_size,
                              hipStream_t stream) {
  const float* x     = (const float*)d_in[0];
  const float* cos_  = (const float*)d_in[1];
  const float* sin_  = (const float*)d_in[2];
  const float* pre_w = (const float*)d_in[3];
  const float* q_w   = (const float*)d_in[4];
  const float* k_w   = (const float*)d_in[5];
  const float* Wq    = (const float*)d_in[6];
  const float* Wk    = (const float*)d_in[7];
  const float* Wv    = (const float*)d_in[8];
  const float* Wproj = (const float*)d_in[9];
  float* out = (float*)d_out;

  char* ws = (char*)d_ws;
  auto B = [&](size_t off) { return (__bf16*)(ws + off); };
  const size_t off_hb   = 0;
  const size_t off_wq   = off_hb   + (size_t)S * H * 2;
  const size_t off_wk   = off_wq   + (size_t)HQ * D * H * 2;
  const size_t off_wv   = off_wk   + (size_t)HKV * D * H * 2;
  const size_t off_wp   = off_wv   + (size_t)HKV * D * H * 2;
  const size_t off_qb   = off_wp   + (size_t)H * HQ * D * 2;
  const size_t off_kb   = off_qb   + (size_t)S * HQ * D * 2;
  const size_t off_vtmp = off_kb   + (size_t)S * HKV * D * 2;
  const size_t off_vt   = off_vtmp + (size_t)S * HKV * D * 2;
  const size_t off_attn = off_vt   + (size_t)S * HKV * D * 2;   // ends ~126MB

  // 1) weights fp32 -> bf16
  {
    int n;
    n = HQ * D * H;  convert_f32_bf16<<<(n + 255) / 256, 256, 0, stream>>>(B(off_wq), Wq, n);
    n = HKV * D * H; convert_f32_bf16<<<(n + 255) / 256, 256, 0, stream>>>(B(off_wk), Wk, n);
    n = HKV * D * H; convert_f32_bf16<<<(n + 255) / 256, 256, 0, stream>>>(B(off_wv), Wv, n);
    n = H * HQ * D;  convert_f32_bf16<<<(n + 255) / 256, 256, 0, stream>>>(B(off_wp), Wproj, n);
  }
  // 2) pre-norm -> hb (bf16)
  rmsnorm_rows<<<S, 256, 0, stream>>>(B(off_hb), x, pre_w);

  // 3) QKV projections (TDM-staged bf16 WMMA, bf16 out)
  gemm_bf16_tdm<true><<<(S / 64) * (HQ * D / 128), 256, 0, stream>>>(
      B(off_qb), B(off_hb), B(off_wq), S, HQ * D, H);
  gemm_bf16_tdm<true><<<(S / 64) * (HKV * D / 128), 256, 0, stream>>>(
      B(off_kb), B(off_hb), B(off_wk), S, HKV * D, H);
  gemm_bf16_tdm<true><<<(S / 64) * (HKV * D / 128), 256, 0, stream>>>(
      B(off_vtmp), B(off_hb), B(off_wv), S, HKV * D, H);

  // 4) per-head RMSNorm + RoPE (in place, bf16)
  rope_norm_inplace<<<S * HQ / 8, 256, 0, stream>>>(B(off_qb), cos_, sin_, q_w, HQ);
  rope_norm_inplace<<<S * HKV / 8, 256, 0, stream>>>(B(off_kb), cos_, sin_, k_w, HKV);

  // 5) V transpose for contiguous P@V fragments
  transpose_v<<<(S * HKV * D) / 256, 256, 0, stream>>>(B(off_vt), B(off_vtmp));

  // 6) flash attention (bf16 WMMA, fp32 accum)
  attn_wmma<<<(HQ * (S / 16)) / 8, 256, 0, stream>>>(B(off_attn), B(off_qb), B(off_kb), B(off_vt));

  // 7) output projection -> fp32 d_out
  gemm_bf16_tdm<false><<<(S / 64) * (H / 128), 256, 0, stream>>>(
      out, B(off_attn), B(off_wp), S, H, HQ * D);
}
